// VectorQuantizer_3169685864517
// MI455X (gfx1250) — matrix operand order repaired and checked by reading, not compiled, not run on hardware
//
#include <hip/hip_runtime.h>
#include <hip/hip_bf16.h>

#define N_ROWS 16384
#define K_CODES 4096
#define D_DIM 512

typedef __attribute__((ext_vector_type(16))) __bf16 v16bf;
typedef __attribute__((ext_vector_type(8)))  float  v8f;
typedef __attribute__((ext_vector_type(4)))  unsigned int u32x4;
typedef __attribute__((ext_vector_type(8)))  int i32x8;
typedef __attribute__((ext_vector_type(4)))  int i32x4;

union FragBF { v16bf v; uint4 q[2]; };

// padded LDS row stride (halfs): 520*2 = 1040 B -> 16B aligned, bank-skewed
#define LDS_STRIDE 520
#define AS_HALFS   (64 * LDS_STRIDE)
#define SLAB_HALFS (64 * LDS_STRIDE)

__device__ __forceinline__ unsigned short f2bf_rne(float f) {
  unsigned u = __builtin_bit_cast(unsigned, f);
  u += 0x7FFFu + ((u >> 16) & 1u);   // round-to-nearest-even
  return (unsigned short)(u >> 16);
}

// ---------------------------------------------------------------------------
// TDM issue: DMA a 64-row x 512-half slab of cb16 into LDS (with row padding
// of 4 DWORDs per 1024B row so the LDS image has stride 520 halfs).
// D# layout per CDNA5 ISA ch.8. Groups 2/3 (and trailing group) zero: 2-D.
// ---------------------------------------------------------------------------
__device__ __forceinline__ void tdm_load_slab(const unsigned short* gsrc,
                                              unsigned lds_byte_addr) {
  const unsigned long long ga = (unsigned long long)(uintptr_t)gsrc;
  u32x4 g0;
  g0.x = 1u;                                            // count=1, user desc
  g0.y = lds_byte_addr;                                 // lds_addr
  g0.z = (unsigned)(ga & 0xFFFFFFFFu);                  // global_addr[31:0]
  g0.w = (unsigned)((ga >> 32) & 0x01FFFFFFu)           // global_addr[56:32]
       | (2u << 30);                                    // type=2 ("image")
  i32x8 g1;
  g1[0] = (int)((1u << 16)        // data_size = 1 -> 2 bytes
              | (1u << 20)        // pad_enable
              | (7u << 22)        // pad_interval: 7 -> every 256 DWORDs (1024B)
              | (3u << 25));      // pad_amount:   3 -> 4 DWORDs (16B)
  g1[1] = (int)(512u << 16);      // tensor_dim0 = 512 (bits 79:48, low half)
  g1[2] = (int)(4096u << 16);     // tensor_dim1 = 4096 (bits 111:80, low half)
  g1[3] = (int)(512u << 16);      // tile_dim0 = 512 (bits 127:112)
  g1[4] = 64;                     // tile_dim1 = 64, tile_dim2 = 0
  g1[5] = 512;                    // tensor_dim0_stride = 512 (bits 207:160)
  g1[6] = 0;                      // dim0_stride hi / dim1_stride lo
  g1[7] = 0;                      // dim1_stride hi
  const i32x4 zz4 = {0, 0, 0, 0};
  const i32x8 zz8 = {0, 0, 0, 0, 0, 0, 0, 0};
  __builtin_amdgcn_tensor_load_to_lds(g0, g1, zz4, zz4, zz8, 0);
}

// ---------------------------------------------------------------------------
// Kernel 1: codebook f32 -> bf16 (L2-resident copy) + per-code ||c||^2
// ---------------------------------------------------------------------------
__global__ void __launch_bounds__(256)
vq_prep_codebook(const float* __restrict__ cb,
                 unsigned short* __restrict__ cb16,
                 float* __restrict__ cnorm) {
  __shared__ float red[256];
  const int k = blockIdx.x;
  const int t = threadIdx.x;
  float s = 0.f;
  for (int d = t; d < D_DIM; d += 256) {
    float v = cb[(size_t)k * D_DIM + d];
    cb16[(size_t)k * D_DIM + d] = f2bf_rne(v);
    s += v * v;
  }
  red[t] = s;
  __syncthreads();
  for (int off = 128; off > 0; off >>= 1) {
    if (t < off) red[t] += red[t + off];
    __syncthreads();
  }
  if (t == 0) cnorm[k] = red[0];
}

// ---------------------------------------------------------------------------
// Kernel 2: WMMA bf16 GEMM (z x cb^T) fused with running argmin over columns.
// ||z||^2 is row-constant -> dropped from the argmin metric.
// Block: 256 threads = 8 waves, tiled 4(M) x 2(N). Block owns 64 rows.
// Codebook slabs (64 cols x 512) arrive via TDM double-buffering in LDS.
// ---------------------------------------------------------------------------
__global__ void __launch_bounds__(256)
vq_argmin_wmma(const float* __restrict__ z,
               const unsigned short* __restrict__ cb16,
               const float* __restrict__ cnorm,
               int* __restrict__ best_idx) {
  extern __shared__ unsigned short smem[];   // As[64][520] | Bs0 | Bs1
  __shared__ float red_d[2][64];
  __shared__ int   red_i[2][64];

  unsigned short* const As  = smem;
  unsigned short* const Bs0 = smem + AS_HALFS;
  unsigned short* const Bs1 = smem + AS_HALFS + SLAB_HALFS;
  const unsigned dynBase = (unsigned)__builtin_amdgcn_groupstaticsize();

  const int tid     = threadIdx.x;
  const int lane    = tid & 31;            // wave32
  const int wid     = tid >> 5;
  const int wave_m  = wid & 3;             // row tile 0..3 (16 rows each)
  const int wave_n  = wid >> 2;            // column half 0..1
  const int rowbase = blockIdx.x * 64;

  // kick off slab 0 DMA before staging A so it overlaps
  if (wid == 0) tdm_load_slab(cb16, dynBase + AS_HALFS * 2u);

  // ---- stage 64x512 rows of z as bf16 into LDS (float4 loads) ----
  for (int i = tid; i < 64 * 128; i += 256) {
    const int r  = i >> 7;
    const int c4 = (i & 127) * 4;
    const float4 zv = *(const float4*)(z + (size_t)(rowbase + r) * D_DIM + c4);
    unsigned short* dst = As + r * LDS_STRIDE + c4;
    dst[0] = f2bf_rne(zv.x);
    dst[1] = f2bf_rne(zv.y);
    dst[2] = f2bf_rne(zv.z);
    dst[3] = f2bf_rne(zv.w);
  }
  if (wid == 0) __builtin_amdgcn_s_wait_tensorcnt(0);
  __syncthreads();   // A staged + slab 0 resident

  const int nlane = lane & 15;
  const int hlane = lane >> 4;

  float bestd[8];
  int   besti[8];
#pragma unroll
  for (int j = 0; j < 8; ++j) { bestd[j] = 3.0e38f; besti[j] = 0; }

  // A-fragment base for this lane: row = wave_m*16 + (lane&15),
  // K offset within a 32-step: +8 for the upper half-wave (ISA 16-bit A layout)
  const unsigned short* paBase =
      As + (wave_m * 16 + nlane) * LDS_STRIDE + hlane * 8;

  for (int it = 0; it < 64; ++it) {
    // prefetch next slab into the other buffer via TDM (EXEC-independent,
    // so only wave 0 executes the instruction)
    if (wid == 0 && it + 1 < 64) {
      const unsigned nextAddr =
          dynBase + (AS_HALFS + (((it + 1) & 1) ? SLAB_HALFS : 0)) * 2u;
      tdm_load_slab(cb16 + (size_t)(it + 1) * 64 * D_DIM, nextAddr);
    }
    const unsigned short* B = (it & 1) ? Bs1 : Bs0;

    v8f acc[2];
#pragma unroll
    for (int t = 0; t < 2; ++t)
#pragma unroll
      for (int j = 0; j < 8; ++j) acc[t][j] = 0.f;

    for (int ks = 0; ks < 16; ++ks) {
      FragBF a;
      const unsigned short* pa = paBase + ks * 32;
      a.q[0] = *(const uint4*)pa;          // K = ks*32 + h*8      .. +7
      a.q[1] = *(const uint4*)(pa + 16);   // K = ks*32 + h*8 + 16 .. +23

      const int bk = ks * 32 + hlane * 8;  // B: lane = column, K = h*8 .. +7, h*8+16 .. +23
#pragma unroll
      for (int t = 0; t < 2; ++t) {
        const int cs = wave_n * 32 + t * 16 + nlane;   // column within slab
        const unsigned short* pb = B + cs * LDS_STRIDE + bk;
        FragBF b;
        b.q[0] = *(const uint4*)pb;
        b.q[1] = *(const uint4*)(pb + 16);
        acc[t] = __builtin_amdgcn_wmma_f32_16x16x32_bf16(
            false, a.v, false, b.v, (short)0, acc[t], false, false);
      }
    }

    // fold this slab into the running argmin
#pragma unroll
    for (int t = 0; t < 2; ++t) {
      const int col = it * 64 + wave_n * 32 + t * 16 + nlane;
      const float cn = cnorm[col];
#pragma unroll
      for (int j = 0; j < 8; ++j) {
        const float d = cn - 2.0f * acc[t][j];
        if (d < bestd[j] || (d == bestd[j] && col < besti[j])) {
          bestd[j] = d;
          besti[j] = col;
        }
      }
    }

    __syncthreads();   // all waves done reading this slab's buffer
    if (wid == 0) __builtin_amdgcn_s_wait_tensorcnt(0);
    __syncthreads();   // next slab resident for everyone
  }

  // ---- reduce across 16 lanes of each half-wave (C-layout: lane = column) ----
#pragma unroll
  for (int j = 0; j < 8; ++j) {
    float d = bestd[j];
    int   i = besti[j];
#pragma unroll
    for (int m = 8; m >= 1; m >>= 1) {
      const float od = __shfl_xor(d, m, 16);
      const int   oi = __shfl_xor(i, m, 16);
      if (od < d || (od == d && oi < i)) { d = od; i = oi; }
    }
    if (nlane == 0) {
      const int r = wave_m * 16 + hlane * 8 + j;
      red_d[wave_n][r] = d;
      red_i[wave_n][r] = i;
    }
  }
  __syncthreads();

  if (tid < 64) {
    const float d0 = red_d[0][tid], d1 = red_d[1][tid];
    const int   i0 = red_i[0][tid], i1 = red_i[1][tid];
    best_idx[rowbase + tid] =
        (d1 < d0 || (d1 == d0 && i1 < i0)) ? i1 : i0;
  }
}

// ---------------------------------------------------------------------------
// Kernel 3: gather z_q in original f32, STE output, loss accumulation
// ---------------------------------------------------------------------------
__global__ void __launch_bounds__(256)
vq_gather_loss(const float* __restrict__ z,
               const float* __restrict__ cb,
               const int* __restrict__ best_idx,
               float* __restrict__ out_zq,
               float* __restrict__ out_idx,
               float* __restrict__ loss_acc) {
  __shared__ float red[256];
  const int row = blockIdx.x;
  const int t   = threadIdx.x;
  const int idx = best_idx[row];
  float s = 0.f;
  for (int d = t; d < D_DIM; d += 256) {
    const float zv = z[(size_t)row * D_DIM + d];
    const float q  = cb[(size_t)idx * D_DIM + d];
    out_zq[(size_t)row * D_DIM + d] = zv + (q - zv);  // bit-exact STE expr
    const float diff = zv - q;
    s += diff * diff;
  }
  red[t] = s;
  __syncthreads();
  for (int off = 128; off > 0; off >>= 1) {
    if (t < off) red[t] += red[t + off];
    __syncthreads();
  }
  if (t == 0) {
    atomicAdd(loss_acc, red[0]);
    out_idx[row] = (float)idx;
  }
}

__global__ void vq_init(float* loss_acc) { *loss_acc = 0.f; }

__global__ void vq_finalize(const float* __restrict__ loss_acc,
                            float* __restrict__ out_loss) {
  // codebook_loss + 0.25*commitment_loss == 1.25 * mean((z - z_q)^2)
  *out_loss = 1.25f * loss_acc[0] / (float)(N_ROWS * D_DIM);
}

// ---------------------------------------------------------------------------
extern "C" void kernel_launch(void* const* d_in, const int* in_sizes, int n_in,
                              void* d_out, int out_size, void* d_ws, size_t ws_size,
                              hipStream_t stream) {
  (void)in_sizes; (void)n_in; (void)out_size; (void)ws_size;
  const float* z  = (const float*)d_in[0];
  const float* cb = (const float*)d_in[1];
  float* out = (float*)d_out;

  char* ws = (char*)d_ws;
  unsigned short* cb16 = (unsigned short*)ws;                       // 4 MB
  float* cnorm    = (float*)(ws + (size_t)K_CODES * D_DIM * 2);     // 16 KB
  int*   best_idx = (int*)((char*)cnorm + (size_t)K_CODES * 4);     // 64 KB
  float* loss_acc = (float*)((char*)best_idx + (size_t)N_ROWS * 4); // 4 B

  vq_init<<<1, 1, 0, stream>>>(loss_acc);
  vq_prep_codebook<<<K_CODES, 256, 0, stream>>>(cb, cb16, cnorm);

  const size_t lds_bytes = (size_t)(AS_HALFS + 2 * SLAB_HALFS) * sizeof(unsigned short);
  vq_argmin_wmma<<<N_ROWS / 64, 256, lds_bytes, stream>>>(z, cb16, cnorm, best_idx);

  vq_gather_loss<<<N_ROWS, 256, 0, stream>>>(
      z, cb, best_idx, out, out + (size_t)N_ROWS * D_DIM, loss_acc);
  vq_finalize<<<1, 1, 0, stream>>>(
      loss_acc, out + (size_t)N_ROWS * D_DIM + N_ROWS);
}
